// Bi_Mamba_block_23167053595294
// MI455X (gfx1250) — matrix operand order repaired and checked by reading, not compiled, not run on hardware
//
#include <hip/hip_runtime.h>
#include <hip/hip_bf16.h>
#include <math.h>

// ---------------- sizes ----------------
#define TB    2
#define TL    1024
#define TE    1024
#define TDI   2048
#define TDS   16
#define TDTR  64
#define NTOK  (TB * TL)      // 2048 tokens
#define KGRID 8
#define KANK  (TE * KGRID)   // 8192

typedef __attribute__((ext_vector_type(16))) _Float16 v16h;
typedef __attribute__((ext_vector_type(8)))  float    v8f;

__device__ __forceinline__ float silu_f(float x) { return x / (1.f + __expf(-x)); }
__device__ __forceinline__ float softplus_f(float x) {
  return (x > 20.f) ? x : log1pf(__expf(x));
}

// ---------------- f32 -> f16 convert ----------------
__global__ __launch_bounds__(256)
void cvt_f32_f16(const float* __restrict__ src, _Float16* __restrict__ dst, long long n) {
  long long i = (long long)blockIdx.x * 256 + threadIdx.x;
  if (i < n) dst[i] = (_Float16)src[i];
}

// ---------------- layernorm (optionally add time-reversed second input) ----
// row = b*TL + l over NTOK rows, 1024 cols; 256 threads/row, 4 cols/thread
__global__ __launch_bounds__(256)
void ln_kernel(const float* __restrict__ a, const float* __restrict__ brev,
               const float* __restrict__ g, const float* __restrict__ beta,
               float* __restrict__ out32, _Float16* __restrict__ out16) {
  __shared__ float red[256];
  const int row = blockIdx.x;
  const int t = threadIdx.x;
  const int bIdx = row >> 10, l = row & 1023;
  const float* pa = a + (size_t)row * TE;
  const float* pb = nullptr;
  if (brev) pb = brev + (((size_t)bIdx << 10) + (TL - 1 - l)) * TE;

  float v[4];
  float s = 0.f;
#pragma unroll
  for (int j = 0; j < 4; ++j) {
    const int c = t + j * 256;
    v[j] = pa[c] + (pb ? pb[c] : 0.f);
    s += v[j];
  }
  red[t] = s; __syncthreads();
  for (int off = 128; off > 0; off >>= 1) {
    if (t < off) red[t] += red[t + off];
    __syncthreads();
  }
  const float mu = red[0] * (1.f / TE);
  __syncthreads();
  float q = 0.f;
#pragma unroll
  for (int j = 0; j < 4; ++j) { const float d = v[j] - mu; q += d * d; }
  red[t] = q; __syncthreads();
  for (int off = 128; off > 0; off >>= 1) {
    if (t < off) red[t] += red[t + off];
    __syncthreads();
  }
  const float rstd = rsqrtf(red[0] * (1.f / TE) + 1e-12f);
#pragma unroll
  for (int j = 0; j < 4; ++j) {
    const int c = t + j * 256;
    const float o = (v[j] - mu) * rstd * g[c] + beta[c];
    if (out32) out32[(size_t)row * TE + c] = o;
    if (out16) out16[(size_t)row * TE + c] = (_Float16)o;
  }
}

// ---------------- causal depthwise conv (DC=4) + bias + silu ----------------
// Reads xz (shared by fwd/bwd); rev!=0 -> operate on time-reversed sequence.
__global__ __launch_bounds__(256)
void conv_silu_kernel(const float* __restrict__ xz, const float* __restrict__ w,
                      const float* __restrict__ bias, float* __restrict__ xc32,
                      _Float16* __restrict__ xc16, int rev) {
  const long long idx = (long long)blockIdx.x * 256 + threadIdx.x; // NTOK*TDI
  const int d = (int)(idx & (TDI - 1));
  const int n = (int)(idx >> 11);
  const int b = n >> 10, l = n & 1023;
  float acc = bias[d];
#pragma unroll
  for (int j = 0; j < 4; ++j) {
    const int tt = l - 3 + j;
    if (tt >= 0) {
      const int src_l = rev ? (TL - 1 - tt) : tt;
      acc += xz[((size_t)(b * TL + src_l)) * (2 * TDI) + d] * w[d * 4 + j];
    }
  }
  const float s = silu_f(acc);
  xc32[idx] = s;
  xc16[idx] = (_Float16)s;
}

// ---------------- extract dt part of x_dbl (cols 0..63) as f16 ----------------
__global__ __launch_bounds__(256)
void extract_dt_kernel(const float* __restrict__ xdbl, _Float16* __restrict__ dtA) {
  const int idx = blockIdx.x * 256 + threadIdx.x;   // NTOK*64
  const int m = idx >> 6, c = idx & 63;
  dtA[idx] = (_Float16)xdbl[(size_t)m * 96 + c];
}

// ---------------- generic NT WMMA GEMM: C[m,n] = sum_k A[m,k]*W[n,k] ----------
// mode 0: store; mode 1: store softplus(acc + bias[n]); mode 2: accumulate.
__global__ __launch_bounds__(256)
void gemm_nt_wmma(const _Float16* __restrict__ A, const _Float16* __restrict__ W,
                  float* __restrict__ C, int M, int N, int K, int mode,
                  const float* __restrict__ bias) {
  constexpr int BM = 128, BN = 64, BK = 32, LDA = 40; // padded LDS stride (halves)
  __shared__ __align__(16) _Float16 As[BM * LDA];
  __shared__ __align__(16) _Float16 Bs[BN * LDA];
  const int tid  = threadIdx.x;
  const int lane = tid & 31;
  const int wv   = tid >> 5;         // 8 waves
  const int wm   = (wv & 3) * 32;    // 4 wave-rows
  const int wn   = (wv >> 2) * 32;   // 2 wave-cols
  const int bm0  = blockIdx.x * BM;
  const int bn0  = blockIdx.y * BN;
  const int lm   = lane & 15;        // M/N index within frag
  const int lh   = lane >> 4;        // K-half select

  v8f acc[2][2] = {};
  union AF { v16h v; unsigned u[8]; };
  union BF { v16h v; int4 q[2]; };

  for (int k0 = 0; k0 < K; k0 += BK) {
    { // A tile: 128 x 32 halves; thread -> row tid>>1, 16 halves
      const int r = tid >> 1, c = (tid & 1) << 4;
      const int4* src = (const int4*)(A + (size_t)(bm0 + r) * K + k0 + c);
      int4* dst = (int4*)(&As[r * LDA + c]);
      dst[0] = src[0];
      dst[1] = src[1];
    }
    { // B tile (= W rows): 64 x 32 halves; thread -> row tid>>2, 8 halves
      const int r = tid >> 2, c = (tid & 3) << 3;
      int4* dst = (int4*)(&Bs[r * LDA + c]);
      const int gn = bn0 + r;
      if (gn < N) *dst = *(const int4*)(W + (size_t)gn * K + k0 + c);
      else        *dst = make_int4(0, 0, 0, 0);
    }
    __syncthreads();

    AF af[2]; BF bf[2];
#pragma unroll
    for (int mf = 0; mf < 2; ++mf) { // A frag per ISA 16-bit 16x32 layout
      const _Float16* row = &As[(wm + mf * 16 + lm) * LDA];
#pragma unroll
      for (int j = 0; j < 4; ++j)
        af[mf].u[j] = *(const unsigned*)&row[lh * 8 + 2 * j];          // K 0..15 half
#pragma unroll
      for (int j = 0; j < 4; ++j)
        af[mf].u[4 + j] = *(const unsigned*)&row[16 + lh * 8 + 2 * j]; // K 16..31 half
    }
#pragma unroll
    for (int nf = 0; nf < 2; ++nf) { // B frag: same K layout as A frag
      const _Float16* row = &Bs[(wn + nf * 16 + lm) * LDA];
      bf[nf].q[0] = *(const int4*)&row[lh * 8];
      bf[nf].q[1] = *(const int4*)&row[16 + lh * 8];
    }
#pragma unroll
    for (int mf = 0; mf < 2; ++mf)
#pragma unroll
      for (int nf = 0; nf < 2; ++nf)
        acc[mf][nf] = __builtin_amdgcn_wmma_f32_16x16x32_f16(
            false, af[mf].v, false, bf[nf].v, (short)0, acc[mf][nf], false, false);
    __syncthreads();
  }

#pragma unroll
  for (int mf = 0; mf < 2; ++mf)
#pragma unroll
    for (int nf = 0; nf < 2; ++nf) {
      const int n = bn0 + wn + nf * 16 + lm;
      if (n >= N) continue;
#pragma unroll
      for (int vv = 0; vv < 8; ++vv) {
        const int m = bm0 + wm + mf * 16 + vv + 8 * lh;
        float val = acc[mf][nf][vv];
        const size_t off = (size_t)m * N + n;
        if (mode == 1)      val = softplus_f(val + bias[n]);
        else if (mode == 2) val += C[off];
        C[off] = val;
      }
    }
}

// ---------------- selective scan: 16 lanes per (b,d), one per state s --------
__global__ __launch_bounds__(256)
void scan_kernel(const float* __restrict__ dt, const float* __restrict__ xc,
                 const float* __restrict__ xdbl, const float* __restrict__ A_log,
                 float* __restrict__ ys) {
  const int tid = blockIdx.x * 256 + threadIdx.x;   // TB*TDI*16 = 65536
  const int s = tid & 15;
  const int grp = tid >> 4;
  const int d = grp & (TDI - 1);
  const int b = grp >> 11;
  const float a = -__expf(A_log[d * TDS + s]);
  float h = 0.f;
  for (int l = 0; l < TL; ++l) {
    const size_t n = (size_t)b * TL + l;
    const float dtv = dt[n * TDI + d];
    const float xcv = xc[n * TDI + d];
    const float bm = xdbl[n * 96 + TDTR + s];
    const float cm = xdbl[n * 96 + TDTR + TDS + s];
    h = h * __expf(dtv * a) + dtv * bm * xcv;
    float p = h * cm;
    p += __shfl_xor(p, 1, 16);
    p += __shfl_xor(p, 2, 16);
    p += __shfl_xor(p, 4, 16);
    p += __shfl_xor(p, 8, 16);
    if (s == 0) ys[n * TDI + d] = p;
  }
}

// ---------------- y = (ys + xc*D) * silu(z) -> f16 ----------------
__global__ __launch_bounds__(256)
void yact_kernel(const float* __restrict__ ys, const float* __restrict__ xc,
                 const float* __restrict__ xz, const float* __restrict__ Dv,
                 _Float16* __restrict__ yact, int rev) {
  const long long idx = (long long)blockIdx.x * 256 + threadIdx.x; // NTOK*TDI
  const int d = (int)(idx & (TDI - 1));
  const int n = (int)(idx >> 11);
  const int b = n >> 10, l = n & 1023;
  const int tz = rev ? (TL - 1 - l) : l;
  const float z = xz[((size_t)(b * TL + tz)) * (2 * TDI) + TDI + d];
  const float y = (ys[idx] + xc[idx] * Dv[d]) * silu_f(z);
  yact[idx] = (_Float16)y;
}

// ---------------- Fourier features: cos/sin(v*k) -> f16 ----------------
__global__ __launch_bounds__(256)
void trig_kernel(const float* __restrict__ v, _Float16* __restrict__ cosF,
                 _Float16* __restrict__ sinF) {
  const int idx = blockIdx.x * 256 + threadIdx.x;   // NTOK*TE
  const float x = v[idx];
  const size_t base = (size_t)idx * KGRID;
#pragma unroll
  for (int g = 0; g < KGRID; ++g) {
    float sn, cs;
    __sincosf(x * (float)(g + 1), &sn, &cs);
    cosF[base + g] = (_Float16)cs;
    sinF[base + g] = (_Float16)sn;
  }
}

// ---------------- final residual ----------------
__global__ __launch_bounds__(256)
void resid_kernel(const float* __restrict__ acc, const float* __restrict__ x,
                  float* __restrict__ out) {
  const int i = blockIdx.x * 256 + threadIdx.x;     // NTOK*TE
  out[i] = acc[i] + x[i];
}

// =======================================================================
extern "C" void kernel_launch(void* const* d_in, const int* in_sizes, int n_in,
                              void* d_out, int out_size, void* d_ws, size_t ws_size,
                              hipStream_t stream) {
  (void)in_sizes; (void)n_in; (void)out_size;
  const size_t MB = 1ull << 20;
  if (ws_size < 141 * MB) return;

  const float* x        = (const float*)d_in[0];
  const float* in_w     = (const float*)d_in[1];
  const float* out_w    = (const float*)d_in[2];
  const float* conv_w[2]= {(const float*)d_in[3],  (const float*)d_in[10]};
  const float* conv_b[2]= {(const float*)d_in[4],  (const float*)d_in[11]};
  const float* xpw[2]   = {(const float*)d_in[5],  (const float*)d_in[12]};
  const float* dtw[2]   = {(const float*)d_in[6],  (const float*)d_in[13]};
  const float* dtb[2]   = {(const float*)d_in[7],  (const float*)d_in[14]};
  const float* Alog[2]  = {(const float*)d_in[8],  (const float*)d_in[15]};
  const float* Dvec[2]  = {(const float*)d_in[9],  (const float*)d_in[16]};
  const float* ln1_g = (const float*)d_in[17], *ln1_b = (const float*)d_in[18];
  const float* ln_g  = (const float*)d_in[19], *ln_b  = (const float*)d_in[20];
  const float* kan_coef = (const float*)d_in[21];
  float* out = (float*)d_out;

  unsigned char* w = (unsigned char*)d_ws;
  float*    xzb    = (float*)(w + 0);          // 32MB  [alias: cosF]
  float*    xc32   = (float*)(w + 32 * MB);    // 16MB  [alias: sinF lo]
  float*    dtbuf  = (float*)(w + 48 * MB);    // 16MB  [alias: sinF hi]
  _Float16* xc16   = (_Float16*)(w + 64 * MB); // 8MB   [alias: coef16]
  float*    ysb    = (float*)(w + 72 * MB);    // 16MB
  _Float16* yact16 = (_Float16*)(w + 88 * MB); // 8MB
  float*    xdbl   = (float*)(w + 96 * MB);    // 0.75MB
  _Float16* dtA16  = (_Float16*)(w + 97 * MB); // 0.25MB
  _Float16* ln116  = (_Float16*)(w + 98 * MB); // 4MB
  float*    outF   = (float*)(w + 102 * MB);   // 8MB   [alias: kan tmp]
  float*    outB   = (float*)(w + 110 * MB);   // 8MB
  float*    ln2b   = (float*)(w + 118 * MB);   // 8MB
  _Float16* inW16  = (_Float16*)(w + 126 * MB);// 8MB
  _Float16* outW16 = (_Float16*)(w + 134 * MB);// 4MB
  _Float16* xpw16[2] = {(_Float16*)(w + 138 * MB), (_Float16*)(w + 138 * MB + 512 * 1024)};
  _Float16* dtw16[2] = {(_Float16*)(w + 139 * MB), (_Float16*)(w + 139 * MB + 512 * 1024)};
  _Float16* cosF   = (_Float16*)(w + 0);       // 32MB over xzb (dead)
  _Float16* sinF   = (_Float16*)(w + 32 * MB); // 32MB over xc32+dtbuf (dead)
  _Float16* coef16 = (_Float16*)(w + 64 * MB); // 32MB over xc16+ysb+yact16 (dead)
  float*    tmpC   = outF;                     // dead after ln2

  auto cvt = [&](const float* s, _Float16* d, long long n) {
    cvt_f32_f16<<<(unsigned)((n + 255) / 256), 256, 0, stream>>>(s, d, n);
  };
  auto gemm = [&](const _Float16* A, const _Float16* Wm, float* C, int M, int N,
                  int K, int mode, const float* bias) {
    dim3 g(M / 128, (N + 63) / 64);
    gemm_nt_wmma<<<g, 256, 0, stream>>>(A, Wm, C, M, N, K, mode, bias);
  };

  // weight conversion f32 -> f16
  cvt(in_w,  inW16,  (long long)2 * TDI * TE);
  cvt(out_w, outW16, (long long)TE * TDI);
  for (int dir = 0; dir < 2; ++dir) {
    cvt(xpw[dir], xpw16[dir], (long long)(TDTR + 2 * TDS) * TDI);
    cvt(dtw[dir], dtw16[dir], (long long)TDI * TDTR);
  }

  // LN1, shared in_proj (reversal = row permutation -> compute once)
  ln_kernel<<<NTOK, 256, 0, stream>>>(x, nullptr, ln1_g, ln1_b, nullptr, ln116);
  gemm(ln116, inW16, xzb, NTOK, 2 * TDI, TE, 0, nullptr);

  for (int dir = 0; dir < 2; ++dir) {
    const int rev = dir;
    conv_silu_kernel<<<(NTOK * TDI) / 256, 256, 0, stream>>>(
        xzb, conv_w[dir], conv_b[dir], xc32, xc16, rev);
    gemm(xc16, xpw16[dir], xdbl, NTOK, TDTR + 2 * TDS, TDI, 0, nullptr);
    extract_dt_kernel<<<(NTOK * TDTR) / 256, 256, 0, stream>>>(xdbl, dtA16);
    gemm(dtA16, dtw16[dir], dtbuf, NTOK, TDI, TDTR, 1, dtb[dir]); // softplus(.+b)
    scan_kernel<<<(TB * TDI * TDS) / 256, 256, 0, stream>>>(
        dtbuf, xc32, xdbl, Alog[dir], ysb);
    yact_kernel<<<(NTOK * TDI) / 256, 256, 0, stream>>>(
        ysb, xc32, xzb, Dvec[dir], yact16, rev);
    gemm(yact16, outW16, dir ? outB : outF, NTOK, TE, TDI, 0, nullptr);
  }

  // LN2 over out_f + time-reversed out_b
  ln_kernel<<<NTOK, 256, 0, stream>>>(outF, outB, ln_g, ln_b, ln2b, nullptr);

  // FourierKAN: features + two NT GEMMs (K = 8192) + residual
  trig_kernel<<<(NTOK * TE) / 256, 256, 0, stream>>>(ln2b, cosF, sinF);
  cvt(kan_coef, coef16, (long long)2 * TE * KANK);
  gemm(cosF, coef16, tmpC, NTOK, TE, KANK, 0, nullptr);
  gemm(sinF, coef16 + (size_t)TE * KANK, tmpC, NTOK, TE, KANK, 2, nullptr);
  resid_kernel<<<(NTOK * TE) / 256, 256, 0, stream>>>(tmpC, x, out);
}
